// SelfAttention_20744692040485
// MI455X (gfx1250) — compile-verified, matrix operand order checked by reading, not run on hardware
//
#include <hip/hip_runtime.h>

// Problem constants (reference: B=4, S=2048, D=1024, H=16, DH=64)
#define BATCH 4
#define S_LEN 2048
#define D_DIM 1024
#define H_NUM 16
#define DH_SZ 64

typedef __attribute__((ext_vector_type(16))) __bf16          v16bf;
typedef __attribute__((ext_vector_type(16))) unsigned short  v16us;
typedef __attribute__((ext_vector_type(8)))  unsigned short  v8us;   // 16 bytes
typedef __attribute__((ext_vector_type(4)))  unsigned short  v4us;   // 8 bytes
typedef __attribute__((ext_vector_type(4)))  float           v4f;    // 16 bytes
typedef __attribute__((ext_vector_type(8)))  float           v8f;
typedef __attribute__((ext_vector_type(4)))  unsigned int    v4ui;
typedef __attribute__((ext_vector_type(8)))  int             v8si;
typedef __attribute__((ext_vector_type(4)))  int             v4si;

union Frag {
  v16bf b;
  v16us u;
  v8us  h[2];
};

// Packed f32 -> bf16 (RNE). Prefer the single-instruction v_cvt_pk_bf16_f32.
#if defined(__has_builtin) && __has_builtin(__builtin_amdgcn_cvt_pk_bf16_f32)
__device__ __forceinline__ unsigned pack2bf(float a, float b) {
  auto t = __builtin_amdgcn_cvt_pk_bf16_f32(a, b);
  unsigned u;
  __builtin_memcpy(&u, &t, sizeof(u));
  return u;
}
#else
__device__ __forceinline__ unsigned pack2bf(float a, float b) {
  union { __bf16 h[2]; unsigned u; } r;
  r.h[0] = (__bf16)a;
  r.h[1] = (__bf16)b;
  return r.u;
}
#endif
__device__ __forceinline__ unsigned short f2bf1(float a) {
  union { __bf16 h; unsigned short u; } r;
  r.h = (__bf16)a;
  return r.u;
}

// 16-bit fragment layout (ISA 7.12.2): lane holds row = lane%16 and its 16
// elements are two contiguous 8-element runs: K = 8*half.. and K = 16+8*half..
// => one fragment = two ds_load_b128 from a row-major tile.
__device__ __forceinline__ v16bf load_row_frag(const unsigned short* p, int stride) {
  const int lane = threadIdx.x & 31;
  const int row = lane & 15, half = lane >> 4;
  const unsigned short* q = p + row * stride + 8 * half;
  Frag f;
  f.h[0] = *(const v8us*)(q);
  f.h[1] = *(const v8us*)(q + 16);
  return f.b;
}

__device__ __forceinline__ v8f wmma_bf16(v16bf a, v16bf b, v8f c) {
  return __builtin_amdgcn_wmma_f32_16x16x32_bf16(false, a, false, b, (short)0, c,
                                                 false, false);
}

// ---------------------------------------------------------------------------
// Kernel 1: fused QKV projection  out = bf16(X @ W + bias)
// grid = (M/64, N/64, 3), block = 128 (4 waves).
// Block tile 64x64; each wave: 16 rows x 64 cols = 4 WMMAs / K-step (A reuse).
// ---------------------------------------------------------------------------
#define LDA 40   // elements; *2 bytes = 80 (multiple of 16)
#define LDB 40

__global__ __launch_bounds__(128) void qkv_gemm_kernel(
    const float* __restrict__ Xq, const float* __restrict__ Xk, const float* __restrict__ Xv,
    const float* __restrict__ Wq, const float* __restrict__ Wk, const float* __restrict__ Wv,
    const float* __restrict__ bq, const float* __restrict__ bk, const float* __restrict__ bv,
    unsigned short* __restrict__ ws) {
  __shared__ __align__(16) unsigned short ldsA[64 * LDA];   // [m][k]
  __shared__ __align__(16) unsigned short ldsBt[64 * LDB];  // [n][k] (transposed)

  const int which = blockIdx.z;
  const float* X    = which == 0 ? Xq : (which == 1 ? Xk : Xv);
  const float* W    = which == 0 ? Wq : (which == 1 ? Wk : Wv);
  const float* bias = which == 0 ? bq : (which == 1 ? bk : bv);
  unsigned short* out = ws + (size_t)which * (size_t)(BATCH * S_LEN) * D_DIM;

  const int tid  = threadIdx.x;
  const int wave = tid >> 5;
  const int lane = tid & 31;
  const int m0   = blockIdx.x * 64;
  const int n0   = blockIdx.y * 64;

  v8f acc[4] = {v8f{}, v8f{}, v8f{}, v8f{}};

  for (int kk = 0; kk < D_DIM; kk += 32) {
    // Stage A tile 64x32 (f32 -> bf16): b128 global loads, b64 LDS stores.
#pragma unroll
    for (int i = 0; i < 4; ++i) {
      int q = tid + 128 * i;          // 512 float4 chunks
      int r = q >> 3;                 // 0..63
      int c = (q & 7) * 4;            // 0..28
      v4f xv = *(const v4f*)&X[(size_t)(m0 + r) * D_DIM + kk + c];
      uint2 pk;
      pk.x = pack2bf(xv[0], xv[1]);
      pk.y = pack2bf(xv[2], xv[3]);
      *(uint2*)&ldsA[r * LDA + c] = pk;
    }
    // Stage W tile 32x64 transposed: each thread owns a 4k x 4n sub-block.
    {
      int r0 = (tid >> 4) * 4;        // k (0..28)
      int c0 = (tid & 15) * 4;        // n (0..60)
      const float* wp = &W[(size_t)(kk + r0) * D_DIM + n0 + c0];
      v4f w0 = *(const v4f*)(wp + 0 * D_DIM);
      v4f w1 = *(const v4f*)(wp + 1 * D_DIM);
      v4f w2 = *(const v4f*)(wp + 2 * D_DIM);
      v4f w3 = *(const v4f*)(wp + 3 * D_DIM);
#pragma unroll
      for (int i = 0; i < 4; ++i) {
        uint2 pk;
        pk.x = pack2bf(w0[i], w1[i]);
        pk.y = pack2bf(w2[i], w3[i]);
        *(uint2*)&ldsBt[(c0 + i) * LDB + r0] = pk;
      }
    }
    if (kk + 32 < D_DIM) {
      __builtin_prefetch(&X[(size_t)(m0 + (tid >> 1)) * D_DIM + kk + 32], 0, 0);
      __builtin_prefetch(&W[(size_t)(kk + 32 + (tid >> 2)) * D_DIM + n0], 0, 0);
    }
    __syncthreads();

    v16bf a = load_row_frag(&ldsA[16 * wave * LDA], LDA);
#pragma unroll
    for (int t = 0; t < 4; ++t) {
      v16bf b = load_row_frag(&ldsBt[16 * t * LDB], LDB);
      acc[t] = wmma_bf16(a, b, acc[t]);
    }
    __syncthreads();
  }

  const int half = lane >> 4, col = lane & 15;
#pragma unroll
  for (int t = 0; t < 4; ++t) {
    float bb = bias[n0 + 16 * t + col];
#pragma unroll
    for (int r = 0; r < 8; ++r) {
      int m = m0 + 16 * wave + r + 8 * half;
      out[(size_t)m * D_DIM + n0 + 16 * t + col] = f2bf1(acc[t][r] + bb);
    }
  }
}

// ---------------------------------------------------------------------------
// Kernel 2: flash attention over bf16 Q/K/V with zero-attention token.
// grid = (S/64, B*H), block = 128 (4 waves; each wave owns 16 query rows).
// 64 keys per iteration: 8 score WMMAs + 8 P@V WMMAs per wave per barrier.
// K tile is fetched by the Tensor Data Mover (tensor_load_to_lds) with LDS
// row padding 128B->144B (pad_interval=32 dwords, pad_amount=4 dwords) so the
// staged stride matches LKV. V staged transposed with packed b64 stores.
// ---------------------------------------------------------------------------
#define LKV  72  // [key 64][dh 64] stride; *2 = 144 bytes (mult of 16)
#define LDVT 72  // [dh 64][key 64] stride

__global__ __launch_bounds__(128) void attn_kernel(
    const unsigned short* __restrict__ Qb,
    const unsigned short* __restrict__ Kb,
    const unsigned short* __restrict__ Vb,
    const float* __restrict__ mask,
    float* __restrict__ out) {
  __shared__ __align__(16) unsigned short ldsK[64 * LKV];    // [key][dh]
  __shared__ __align__(16) unsigned short ldsVt[64 * LDVT];  // [dh][key]
  __shared__ __align__(16) unsigned short ldsP[4][16 * 64];  // per-wave P

  const int tid  = threadIdx.x;
  const int wave = tid >> 5;
  const int lane = tid & 31;
  const int half = lane >> 4;
  const int col  = lane & 15;

  const int b  = blockIdx.y >> 4;
  const int h  = blockIdx.y & 15;
  const int q0 = blockIdx.x * 64 + wave * 16;

  const size_t rowbase = (size_t)b * S_LEN;
  const int hoff = h * DH_SZ;

  // Preload Q fragments with 4 global b128 loads (contiguous dh runs).
  const unsigned short* qp = &Qb[(rowbase + q0 + col) * D_DIM + hoff + 8 * half];
  Frag fq0, fq1;
  fq0.h[0] = *(const v8us*)(qp);
  fq0.h[1] = *(const v8us*)(qp + 16);
  fq1.h[0] = *(const v8us*)(qp + 32);
  fq1.h[1] = *(const v8us*)(qp + 48);
  const v16bf aq0 = fq0.b, aq1 = fq1.b;

  // TDM descriptor group 1 (constant across iterations):
  //   data_size=1 (2 bytes)           -> bits[17:16]
  //   pad_enable=1                    -> bit[20]
  //   pad_interval=4 (32 dwords=128B) -> bits[24:22]
  //   pad_amount=3 (4 dwords=16B)     -> bits[31:25]
  //   tensor_dim0=1024, tensor_dim1=BATCH*S_LEN, tile=64x64, stride0=1024
  v8si tg1;
  tg1[0] = (int)((1u << 16) | (1u << 20) | (4u << 22) | (3u << 25));
  tg1[1] = (int)(((unsigned)D_DIM & 0xFFFFu) << 16);             // tensor_dim0 lo
  tg1[2] = (int)((((unsigned)D_DIM >> 16) & 0xFFFFu) |
                 (((unsigned)(BATCH * S_LEN) & 0xFFFFu) << 16)); // dim0 hi | dim1 lo
  tg1[3] = (int)(((unsigned)(BATCH * S_LEN) >> 16) | (64u << 16)); // dim1 hi | tile_dim0
  tg1[4] = 64;                                                   // tile_dim1 (tile_dim2=0)
  tg1[5] = D_DIM;                                                // tensor_dim0_stride lo
  tg1[6] = 0;
  tg1[7] = 0;
  const unsigned ldsKaddr = (unsigned)(unsigned long long)&ldsK[0];
  const v4si zg4 = {0, 0, 0, 0};
  const v8si zg8 = {0, 0, 0, 0, 0, 0, 0, 0};

  const float* mrow = &mask[(size_t)b * (S_LEN + 1)];
  const float m0v = mrow[0];  // zero-token masked score
  float mr[8], lr[8];
#pragma unroll
  for (int r = 0; r < 8; ++r) { mr[r] = m0v; lr[r] = 1.0f; }
  v8f o0 = {}, o1 = {}, o2 = {}, o3 = {};

  const float scale = 0.125f;  // 1/sqrt(64)

  for (int kb = 0; kb < S_LEN; kb += 64) {
    // Stage K 64x64 bf16 tile with the Tensor Data Mover (wave 0 issues).
    if (wave == 0) {
      unsigned long long gaddr =
          (unsigned long long)&Kb[(rowbase + kb) * D_DIM + hoff];
      v4ui tg0;
      tg0[0] = 1u;                                   // count=1, user descriptor
      tg0[1] = ldsKaddr;                             // lds_addr
      tg0[2] = (unsigned)(gaddr & 0xFFFFFFFFull);    // global_addr lo
      tg0[3] = (unsigned)((gaddr >> 32) & 0x1FFFFFFull) | (2u << 30);  // hi|type=2
      __builtin_amdgcn_tensor_load_to_lds(tg0, tg1, zg4, zg4, zg8, 0);
      __builtin_amdgcn_s_wait_tensorcnt(0);
    }
    // Stage V transposed: 4key x 4dh sub-blocks, packed b64 stores.
#pragma unroll
    for (int i = 0; i < 2; ++i) {
      int sb   = tid + 128 * i;      // 256 sub-blocks
      int key0 = (sb & 15) * 4;
      int dh0  = (sb >> 4) * 4;
      const unsigned short* vp = &Vb[(rowbase + kb + key0) * D_DIM + hoff + dh0];
      v4us v0 = *(const v4us*)(vp + 0 * D_DIM);
      v4us v1 = *(const v4us*)(vp + 1 * D_DIM);
      v4us v2 = *(const v4us*)(vp + 2 * D_DIM);
      v4us v3 = *(const v4us*)(vp + 3 * D_DIM);
#pragma unroll
      for (int j = 0; j < 4; ++j) {
        uint2 pk;
        pk.x = (unsigned)v0[j] | ((unsigned)v1[j] << 16);
        pk.y = (unsigned)v2[j] | ((unsigned)v3[j] << 16);
        *(uint2*)&ldsVt[(dh0 + j) * LDVT + key0] = pk;
      }
    }
    __syncthreads();

    // Scores: four 16x16 key tiles, K-dim = DH (2 x 32) each.
    v8f s0 = {}, s1 = {}, s2 = {}, s3 = {};
    s0 = wmma_bf16(aq0, load_row_frag(&ldsK[0 * LKV + 0],  LKV), s0);
    s0 = wmma_bf16(aq1, load_row_frag(&ldsK[0 * LKV + 32], LKV), s0);
    s1 = wmma_bf16(aq0, load_row_frag(&ldsK[16 * LKV + 0],  LKV), s1);
    s1 = wmma_bf16(aq1, load_row_frag(&ldsK[16 * LKV + 32], LKV), s1);
    s2 = wmma_bf16(aq0, load_row_frag(&ldsK[32 * LKV + 0],  LKV), s2);
    s2 = wmma_bf16(aq1, load_row_frag(&ldsK[32 * LKV + 32], LKV), s2);
    s3 = wmma_bf16(aq0, load_row_frag(&ldsK[48 * LKV + 0],  LKV), s3);
    s3 = wmma_bf16(aq1, load_row_frag(&ldsK[48 * LKV + 32], LKV), s3);

    const float mv0 = mrow[1 + kb + col];
    const float mv1 = mrow[1 + kb + 16 + col];
    const float mv2 = mrow[1 + kb + 32 + col];
    const float mv3 = mrow[1 + kb + 48 + col];

    unsigned short* pp = &ldsP[wave][0];
#pragma unroll
    for (int r = 0; r < 8; ++r) {
      // Row r of the accumulators lives in a 16-lane half; xor<16 reductions.
      float x0 = s0[r] * scale + mv0;
      float x1 = s1[r] * scale + mv1;
      float x2 = s2[r] * scale + mv2;
      float x3 = s3[r] * scale + mv3;
      float mx = fmaxf(fmaxf(x0, x1), fmaxf(x2, x3));
      mx = fmaxf(mx, __shfl_xor(mx, 1));
      mx = fmaxf(mx, __shfl_xor(mx, 2));
      mx = fmaxf(mx, __shfl_xor(mx, 4));
      mx = fmaxf(mx, __shfl_xor(mx, 8));
      float mnew  = fmaxf(mr[r], mx);
      float alpha = __expf(mr[r] - mnew);
      float p0 = __expf(x0 - mnew);
      float p1 = __expf(x1 - mnew);
      float p2 = __expf(x2 - mnew);
      float p3 = __expf(x3 - mnew);
      float rs = (p0 + p1) + (p2 + p3);
      rs += __shfl_xor(rs, 1);
      rs += __shfl_xor(rs, 2);
      rs += __shfl_xor(rs, 4);
      rs += __shfl_xor(rs, 8);
      lr[r] = lr[r] * alpha + rs;
      mr[r] = mnew;
      o0[r] *= alpha; o1[r] *= alpha; o2[r] *= alpha; o3[r] *= alpha;
      // C-layout (row r+8*half, col lane%16) -> row-major P tile [16][64].
      int rowM = r + 8 * half;
      unsigned short* pr = pp + rowM * 64 + col;
      pr[0]  = f2bf1(p0);
      pr[16] = f2bf1(p1);
      pr[32] = f2bf1(p2);
      pr[48] = f2bf1(p3);
    }

    // P (16x64) @ V (64x64): 2 A-fragments (keys 0..31 / 32..63), 8 WMMAs.
    v16bf ap0 = load_row_frag(pp, 64);
    v16bf ap1 = load_row_frag(pp + 32, 64);
    o0 = wmma_bf16(ap0, load_row_frag(&ldsVt[0 * LDVT],       LDVT), o0);
    o0 = wmma_bf16(ap1, load_row_frag(&ldsVt[0 * LDVT + 32],  LDVT), o0);
    o1 = wmma_bf16(ap0, load_row_frag(&ldsVt[16 * LDVT],      LDVT), o1);
    o1 = wmma_bf16(ap1, load_row_frag(&ldsVt[16 * LDVT + 32], LDVT), o1);
    o2 = wmma_bf16(ap0, load_row_frag(&ldsVt[32 * LDVT],      LDVT), o2);
    o2 = wmma_bf16(ap1, load_row_frag(&ldsVt[32 * LDVT + 32], LDVT), o2);
    o3 = wmma_bf16(ap0, load_row_frag(&ldsVt[48 * LDVT],      LDVT), o3);
    o3 = wmma_bf16(ap1, load_row_frag(&ldsVt[48 * LDVT + 32], LDVT), o3);
    __syncthreads();
  }

#pragma unroll
  for (int r = 0; r < 8; ++r) {
    float inv = 1.0f / lr[r];
    size_t base = (rowbase + q0 + r + 8 * half) * D_DIM + hoff + col;
    out[base + 0]  = o0[r] * inv;
    out[base + 16] = o1[r] * inv;
    out[base + 32] = o2[r] * inv;
    out[base + 48] = o3[r] * inv;
  }
}

// ---------------------------------------------------------------------------
extern "C" void kernel_launch(void* const* d_in, const int* in_sizes, int n_in,
                              void* d_out, int out_size, void* d_ws, size_t ws_size,
                              hipStream_t stream) {
  const float* Xq   = (const float*)d_in[0];
  const float* Xk   = (const float*)d_in[1];
  const float* Xv   = (const float*)d_in[2];
  const float* mask = (const float*)d_in[3];
  const float* Wq   = (const float*)d_in[4];
  const float* bq   = (const float*)d_in[5];
  const float* Wk   = (const float*)d_in[6];
  const float* bk   = (const float*)d_in[7];
  const float* Wv   = (const float*)d_in[8];
  const float* bv   = (const float*)d_in[9];
  unsigned short* ws = (unsigned short*)d_ws;  // bf16 Q|K|V, 3 x 16 MB

  dim3 g1((BATCH * S_LEN) / 64, D_DIM / 64, 3);
  qkv_gemm_kernel<<<g1, 128, 0, stream>>>(Xq, Xk, Xv, Wq, Wk, Wv, bq, bk, bv, ws);

  const unsigned short* Qb = ws;
  const unsigned short* Kb = ws + (size_t)1 * BATCH * S_LEN * D_DIM;
  const unsigned short* Vb = ws + (size_t)2 * BATCH * S_LEN * D_DIM;
  dim3 g2(S_LEN / 64, BATCH * H_NUM);
  attn_kernel<<<g2, 128, 0, stream>>>(Qb, Kb, Vb, mask, (float*)d_out);
}
